// HypothesisScorer_3959959847188
// MI455X (gfx1250) — compile-verified, matrix operand order checked by reading, not run on hardware
//
#include <hip/hip_runtime.h>
#include <cstdint>
#include <cstddef>

#define NN   50000
#define NE   800000
#define HID  128
#define KPAD 160
#define NHYP 8192
#define EPSV 1e-5f

typedef _Float16 h16;
typedef __attribute__((ext_vector_type(4)))  _Float16 v4h;
typedef __attribute__((ext_vector_type(8)))  _Float16 v8h;
typedef __attribute__((ext_vector_type(16))) _Float16 v16h;
typedef __attribute__((ext_vector_type(8)))  float    v8f;

static __device__ inline v16h cat16(v8h a, v8h b) {
  return __builtin_shufflevector(a, b, 0,1,2,3,4,5,6,7,8,9,10,11,12,13,14,15);
}
static __device__ inline unsigned enc_key(float f) {
  unsigned u = __float_as_uint(f);
  return (u & 0x80000000u) ? ~u : (u | 0x80000000u);
}
static __device__ inline float dec_key(unsigned k) {
  return (k & 0x80000000u) ? __uint_as_float(k & 0x7fffffffu) : __uint_as_float(~k);
}
static __device__ inline float wave_sum(float v) {
  for (int m = 16; m; m >>= 1) v += __shfl_xor(v, m, 32);
  return v;
}
#define LDS_FENCE() asm volatile("s_wait_dscnt 0" ::: "memory")

// ---------------------------------------------------------------- init
__global__ void __launch_bounds__(256) k_init(float* aggr, float* denom,
                                              unsigned* amax, int* counter) {
  int t = blockIdx.x * blockDim.x + threadIdx.x;
  if (t < NN * HID) aggr[t] = 0.f;
  if (t < NN) { denom[t] = 0.f; amax[t] = 0u; }
  if (t == 0) counter[0] = 0;
}

// ---------------------------------------------------------------- edge compaction
__global__ void __launch_bounds__(256) k_compact(const int* ei, const int* ntype,
                                                 int* counter, int* src_act,
                                                 int* dst_act, int* eid_act) {
  int e = blockIdx.x * blockDim.x + threadIdx.x;
  if (e >= NE) return;
  int s = ei[e], d = ei[NE + e];
  if (ntype[s] == 0 && ntype[d] == 1) {
    int slot = atomicAdd(counter, 1);
    src_act[slot] = s; dst_act[slot] = d; eid_act[slot] = e;
  }
}

// ---------------------------------------------------------------- weight prep (transpose+pad to f16)
__global__ void __launch_bounds__(256) k_prep(const float* mw1, const float* aw1,
                                              const float* mw2, h16* mw1t,
                                              h16* aw1t, h16* mw2t) {
  int t = blockIdx.x * blockDim.x + threadIdx.x;
  if (t < 128 * KPAD) {                       // msg_w1 [131,128] -> [128][160]
    int n = t / KPAD, k = t % KPAD;
    mw1t[t] = (k < 131) ? (h16)mw1[k * 128 + n] : (h16)0.f;
  } else if (t < 128 * KPAD + 64 * KPAD) {    // att_w1 [131,64] -> [64][160]
    int q = t - 128 * KPAD; int n = q / KPAD, k = q % KPAD;
    aw1t[q] = (k < 131) ? (h16)aw1[k * 64 + n] : (h16)0.f;
  } else if (t < 128 * KPAD + 64 * KPAD + 128 * 128) { // msg_w2 [128,128] -> [128][128]
    int q = t - 128 * KPAD - 64 * KPAD; int n = q / 128, k = q % 128;
    mw2t[q] = (h16)mw2[k * 128 + n];
  }
}

// ---------------------------------------------------------------- encoder (tiny GEMMs, VALU)
__global__ void __launch_bounds__(256) k_encoder(const float* x, const float* w1,
                                                 const float* b1, const float* w2,
                                                 const float* b2, float* h) {
  __shared__ float t1[8][HID];
  int lane = threadIdx.x & 31, wv = threadIdx.x >> 5;
  int gw = blockIdx.x * 8 + wv, nw = gridDim.x * 8;
  for (int node = gw; node < NN; node += nw) {
    float xv[6];
    for (int k = 0; k < 6; ++k) xv[k] = x[node * 6 + k];
    for (int j = lane; j < HID; j += 32) {
      float acc = b1[j];
      for (int k = 0; k < 6; ++k) acc += xv[k] * w1[k * HID + j];
      t1[wv][j] = fmaxf(acc, 0.f);
    }
    LDS_FENCE();
    for (int j = lane; j < HID; j += 32) {
      float acc = b2[j];
      for (int k = 0; k < HID; ++k) acc += t1[wv][k] * w2[k * HID + j];
      h[(size_t)node * HID + j] = fmaxf(acc, 0.f);
    }
    LDS_FENCE();
  }
}

// ---------------------------------------------------------------- stage a 16-edge z tile into LDS
static __device__ inline void stage_ztile(h16* zbase, const float* h,
                                          const float* edge_attr, const int* sS,
                                          const int* sE, int lane) {
  for (int r = 0; r < 16; ++r) {
    h16* zr = zbase + r * KPAD;
    int s = sS[r];
    v4h q;
    if (s >= 0) {
      const float4 f = ((const float4*)(h + (size_t)s * HID))[lane];
      q[0] = (h16)f.x; q[1] = (h16)f.y; q[2] = (h16)f.z; q[3] = (h16)f.w;
    } else {
      q[0] = (h16)0.f; q[1] = (h16)0.f; q[2] = (h16)0.f; q[3] = (h16)0.f;
    }
    *((v4h*)(zr + lane * 4)) = q;        // halves 0..127
    int e = sE[r];
    h16 tail = (h16)0.f;
    if (lane < 3 && e >= 0) tail = (h16)edge_attr[(size_t)e * 3 + lane];
    zr[128 + lane] = tail;               // halves 128..159 (attr + zero pad)
  }
}

// A/B fragment load helper: base = row-major K-stride array, sel = lane>=16
static __device__ inline v16h load_frag(const h16* p) {
  v8h lo = *(const v8h*)p;
  v8h hi = *(const v8h*)(p + 16);
  return cat16(lo, hi);
}

// ---------------------------------------------------------------- attention pass (WMMA)
__global__ void __launch_bounds__(256) k_att(const float* h, const float* edge_attr,
                                             const int* src_act, const int* dst_act,
                                             const int* eid_act, const int* counter,
                                             const h16* aw1t, const float* ab1,
                                             const float* aw2, const float* ab2,
                                             float* a_ws, unsigned* amax) {
  __shared__ __align__(16) h16 ztile[8][16 * KPAD];
  __shared__ int sS[8][16], sD[8][16], sE[8][16];
  int lane = threadIdx.x & 31, wv = threadIdx.x >> 5;
  int gw = blockIdx.x * 8 + wv, nw = gridDim.x * 8;
  int nact = counter[0];
  int ntiles = (nact + 15) >> 4;
  int mrow = lane & 15, sel = lane >> 4;

  for (int tile = gw; tile < ntiles; tile += nw) {
    int base = tile * 16;
    if (lane < 16) {
      int slot = base + lane;
      bool v = slot < nact;
      sS[wv][lane] = v ? src_act[slot] : -1;
      sD[wv][lane] = v ? dst_act[slot] : -1;
      sE[wv][lane] = v ? eid_act[slot] : -1;
    }
    LDS_FENCE();
    stage_ztile(&ztile[wv][0], h, edge_attr, sS[wv], sE[wv], lane);
    LDS_FENCE();

    v8f acc[4];
    v8f z8 = {0.f,0.f,0.f,0.f,0.f,0.f,0.f,0.f};
    for (int n = 0; n < 4; ++n) acc[n] = z8;
    for (int c = 0; c < 5; ++c) {
      v16h af = load_frag(&ztile[wv][mrow * KPAD + c * 32 + sel * 8]);
      for (int n = 0; n < 4; ++n) {
        v16h bf = load_frag(aw1t + (size_t)(n * 16 + mrow) * KPAD + c * 32 + sel * 8);
        acc[n] = __builtin_amdgcn_wmma_f32_16x16x32_f16(false, af, false, bf,
                                                        (short)0, acc[n], false, false);
      }
    }
    // bias + relu -> f16 staging (reuse z region, stride 64)
    h16* ah = &ztile[wv][0];
    for (int n = 0; n < 4; ++n) {
      int col = n * 16 + mrow;
      float bb = ab1[col];
      for (int r = 0; r < 8; ++r) {
        int row = r + sel * 8;
        ah[row * 64 + col] = (h16)fmaxf(acc[n][r] + bb, 0.f);
      }
    }
    LDS_FENCE();
    if (lane < 16) {
      int slot = base + lane;
      if (slot < nact) {
        float a = ab2[0];
        for (int j = 0; j < 64; ++j) a += (float)ah[lane * 64 + j] * aw2[j];
        a_ws[slot] = a;
        atomicMax(amax + sD[wv][lane], enc_key(a));
      }
    }
    LDS_FENCE();
  }
}

// ---------------------------------------------------------------- message pass (WMMA) + scatter
__global__ void __launch_bounds__(256) k_msg(const float* h, const float* edge_attr,
                                             const int* src_act, const int* dst_act,
                                             const int* eid_act, const int* counter,
                                             const h16* mw1t, const float* mb1,
                                             const h16* mw2t, const float* mb2,
                                             const float* a_ws, const unsigned* amax,
                                             float* denom, float* aggr) {
  __shared__ __align__(16) h16 ztile[8][16 * KPAD];
  __shared__ int sS[8][16], sD[8][16], sE[8][16];
  __shared__ float sW[8][16];
  int lane = threadIdx.x & 31, wv = threadIdx.x >> 5;
  int gw = blockIdx.x * 8 + wv, nw = gridDim.x * 8;
  int nact = counter[0];
  int ntiles = (nact + 15) >> 4;
  int mrow = lane & 15, sel = lane >> 4;

  for (int tile = gw; tile < ntiles; tile += nw) {
    int base = tile * 16;
    if (lane < 16) {
      int slot = base + lane;
      bool v = slot < nact;
      sS[wv][lane] = v ? src_act[slot] : -1;
      sD[wv][lane] = v ? dst_act[slot] : -1;
      sE[wv][lane] = v ? eid_act[slot] : -1;
    }
    LDS_FENCE();
    stage_ztile(&ztile[wv][0], h, edge_attr, sS[wv], sE[wv], lane);
    LDS_FENCE();

    // GEMM1: z[16x160] @ W1t -> hidden[16x128]
    v8f acc[8];
    v8f z8 = {0.f,0.f,0.f,0.f,0.f,0.f,0.f,0.f};
    for (int n = 0; n < 8; ++n) acc[n] = z8;
    for (int c = 0; c < 5; ++c) {
      v16h af = load_frag(&ztile[wv][mrow * KPAD + c * 32 + sel * 8]);
      for (int n = 0; n < 8; ++n) {
        v16h bf = load_frag(mw1t + (size_t)(n * 16 + mrow) * KPAD + c * 32 + sel * 8);
        acc[n] = __builtin_amdgcn_wmma_f32_16x16x32_f16(false, af, false, bf,
                                                        (short)0, acc[n], false, false);
      }
    }
    // bias + relu -> f16 hidden tile in LDS (stride 128), C-layout -> memory
    h16* hd = &ztile[wv][0];
    for (int n = 0; n < 8; ++n) {
      int col = n * 16 + mrow;
      float bb = mb1[col];
      for (int r = 0; r < 8; ++r) {
        int row = r + sel * 8;
        hd[row * 128 + col] = (h16)fmaxf(acc[n][r] + bb, 0.f);
      }
    }
    LDS_FENCE();
    // softmax weights (un-normalized; denom accumulated for later division)
    if (lane < 16) {
      int slot = base + lane;
      float w = 0.f;
      if (slot < nact) {
        int d = sD[wv][lane];
        float am = dec_key(amax[d]);
        w = __expf(a_ws[slot] - am);
        atomicAdd(denom + d, w);
      }
      sW[wv][lane] = w;
    }
    LDS_FENCE();

    // GEMM2: hidden[16x128] @ W2t -> m[16x128]
    v8f acc2[8];
    for (int n = 0; n < 8; ++n) acc2[n] = z8;
    for (int c = 0; c < 4; ++c) {
      v16h af = load_frag(hd + mrow * 128 + c * 32 + sel * 8);
      for (int n = 0; n < 8; ++n) {
        v16h bf = load_frag(mw2t + (size_t)(n * 16 + mrow) * 128 + c * 32 + sel * 8);
        acc2[n] = __builtin_amdgcn_wmma_f32_16x16x32_f16(false, af, false, bf,
                                                         (short)0, acc2[n], false, false);
      }
    }
    float wr[8]; int dr[8];
    for (int r = 0; r < 8; ++r) {
      int row = r + sel * 8;
      wr[r] = sW[wv][row];
      dr[r] = sD[wv][row];
    }
    for (int n = 0; n < 8; ++n) {
      int col = n * 16 + mrow;
      float bb = mb2[col];
      for (int r = 0; r < 8; ++r) {
        if (dr[r] >= 0)
          atomicAdd(aggr + (size_t)dr[r] * HID + col, (acc2[n][r] + bb) * wr[r]);
      }
    }
    LDS_FENCE();
  }
}

// ---------------------------------------------------------------- node update: LN + relu + residual
__global__ void __launch_bounds__(256) k_node(float* h, float* aggr, float* denom,
                                              unsigned* amax, const float* g,
                                              const float* b) {
  int lane = threadIdx.x & 31, wv = threadIdx.x >> 5;
  int gw = blockIdx.x * 8 + wv, nw = gridDim.x * 8;
  for (int node = gw; node < NN; node += nw) {
    const float4 hv4 = ((const float4*)(h + (size_t)node * HID))[lane];
    const float4 av4 = ((const float4*)(aggr + (size_t)node * HID))[lane];
    float dn = denom[node];
    float scale = (dn > 0.f) ? (1.f / dn) : 1.f;
    float hv[4] = {hv4.x, hv4.y, hv4.z, hv4.w};
    float av[4] = {av4.x, av4.y, av4.z, av4.w};
    float v[4];
    float s = 0.f;
    for (int i = 0; i < 4; ++i) { v[i] = hv[i] + av[i] * scale; s += v[i]; }
    s = wave_sum(s);
    float mu = s * (1.f / HID);
    float q = 0.f;
    for (int i = 0; i < 4; ++i) { float d = v[i] - mu; q += d * d; }
    q = wave_sum(q);
    float rs = rsqrtf(q * (1.f / HID) + EPSV);
    float4 out;
    float* op = &out.x;
    for (int i = 0; i < 4; ++i) {
      int k = lane * 4 + i;
      op[i] = fmaxf((v[i] - mu) * rs * g[k] + b[k], 0.f) + hv[i];
    }
    ((float4*)(h + (size_t)node * HID))[lane] = out;
    float4 zero4 = {0.f, 0.f, 0.f, 0.f};
    ((float4*)(aggr + (size_t)node * HID))[lane] = zero4;   // reset for next layer
    if (lane == 0) { denom[node] = 0.f; amax[node] = 0u; }
  }
}

// ---------------------------------------------------------------- head MLP
__global__ void __launch_bounds__(256) k_head(const float* h, const int* hyp,
                                              const float* w1, const float* b1,
                                              const float* w2, const float* b2,
                                              float* out) {
  __shared__ float hrow[8][HID];
  int lane = threadIdx.x & 31, wv = threadIdx.x >> 5;
  int gw = blockIdx.x * 8 + wv, nw = gridDim.x * 8;
  for (int i = gw; i < NHYP; i += nw) {
    int node = hyp[i];
    const float4 f = ((const float4*)(h + (size_t)node * HID))[lane];
    hrow[wv][lane * 4 + 0] = f.x; hrow[wv][lane * 4 + 1] = f.y;
    hrow[wv][lane * 4 + 2] = f.z; hrow[wv][lane * 4 + 3] = f.w;
    LDS_FENCE();
    float a0 = b1[lane], a1 = b1[lane + 32];
    for (int k = 0; k < HID; ++k) {
      float v = hrow[wv][k];
      a0 += v * w1[k * 64 + lane];
      a1 += v * w1[k * 64 + lane + 32];
    }
    float p = fmaxf(a0, 0.f) * w2[lane] + fmaxf(a1, 0.f) * w2[lane + 32];
    p = wave_sum(p);
    if (lane == 0) out[i] = p + b2[0];
    LDS_FENCE();
  }
}

// ================================================================ host
extern "C" void kernel_launch(void* const* d_in, const int* in_sizes, int n_in,
                              void* d_out, int out_size, void* d_ws, size_t ws_size,
                              hipStream_t stream) {
  (void)n_in; (void)out_size; (void)ws_size;
  // --- detect pytree flatten order: insertion (x first) vs sorted (edge_attr first)
  bool insertion = (in_sizes[0] == NN * 6);
  int I_x, I_ei, I_ea, I_nt, I_hy, I_ew1, I_eb1, I_ew2, I_eb2;
  int I_mw1[3], I_mb1[3], I_mw2[3], I_mb2[3];
  int I_aw1[3], I_ab1[3], I_aw2[3], I_ab2[3];
  int I_lg[3], I_lb[3], I_hw1, I_hb1, I_hw2, I_hb2;
  if (insertion) {
    I_x = 0; I_ei = 1; I_ea = 2; I_nt = 3; I_hy = 4;
    I_ew1 = 5; I_eb1 = 6; I_ew2 = 7; I_eb2 = 8;
    for (int l = 0; l < 3; ++l) {
      int base = 9 + 10 * l;
      I_mw1[l] = base;     I_mb1[l] = base + 1;
      I_mw2[l] = base + 2; I_mb2[l] = base + 3;
      I_aw1[l] = base + 4; I_ab1[l] = base + 5;
      I_aw2[l] = base + 6; I_ab2[l] = base + 7;
      I_lg[l]  = base + 8; I_lb[l]  = base + 9;
    }
    I_hw1 = 39; I_hb1 = 40; I_hw2 = 41; I_hb2 = 42;
  } else {
    I_ea = 0; I_ei = 1; I_hy = 2; I_nt = 3;
    for (int l = 0; l < 3; ++l) {
      I_ab1[l] = 4 + l;  I_ab2[l] = 7 + l;
      I_aw1[l] = 10 + l; I_aw2[l] = 13 + l;
    }
    I_eb1 = 16; I_eb2 = 17; I_ew1 = 18; I_ew2 = 19;
    I_hb1 = 20; I_hb2 = 21; I_hw1 = 22; I_hw2 = 23;
    for (int l = 0; l < 3; ++l) {
      I_lb[l]  = 24 + l; I_lg[l]  = 27 + l;
      I_mb1[l] = 30 + l; I_mb2[l] = 33 + l;
      I_mw1[l] = 36 + l; I_mw2[l] = 39 + l;
    }
    I_x = 42;
  }
  const float* X  = (const float*)d_in[I_x];
  const int*   EI = (const int*)d_in[I_ei];
  const float* EA = (const float*)d_in[I_ea];
  const int*   NT = (const int*)d_in[I_nt];
  const int*   HY = (const int*)d_in[I_hy];

  // --- workspace partition
  char* ws = (char*)d_ws;
  size_t off = 0;
  auto alloc = [&](size_t bytes) -> void* {
    void* p = ws + off;
    off = (off + bytes + 255) & ~(size_t)255;
    return p;
  };
  float*    aggr    = (float*)alloc((size_t)NN * HID * 4);
  float*    hbuf    = (float*)alloc((size_t)NN * HID * 4);
  float*    denom   = (float*)alloc((size_t)NN * 4);
  unsigned* amax    = (unsigned*)alloc((size_t)NN * 4);
  int*      counter = (int*)alloc(256);
  int*      src_act = (int*)alloc((size_t)NE * 4);
  int*      dst_act = (int*)alloc((size_t)NE * 4);
  int*      eid_act = (int*)alloc((size_t)NE * 4);
  float*    a_ws    = (float*)alloc((size_t)NE * 4);
  h16* mw1t[3]; h16* aw1t[3]; h16* mw2t[3];
  for (int l = 0; l < 3; ++l) {
    mw1t[l] = (h16*)alloc((size_t)128 * KPAD * 2);
    aw1t[l] = (h16*)alloc((size_t)64 * KPAD * 2);
    mw2t[l] = (h16*)alloc((size_t)128 * 128 * 2);
  }

  // --- pipeline
  k_init<<<(NN * HID + 255) / 256, 256, 0, stream>>>(aggr, denom, amax, counter);
  for (int l = 0; l < 3; ++l) {
    k_prep<<<184, 256, 0, stream>>>((const float*)d_in[I_mw1[l]],
                                    (const float*)d_in[I_aw1[l]],
                                    (const float*)d_in[I_mw2[l]],
                                    mw1t[l], aw1t[l], mw2t[l]);
  }
  k_encoder<<<256, 256, 0, stream>>>(X, (const float*)d_in[I_ew1],
                                     (const float*)d_in[I_eb1],
                                     (const float*)d_in[I_ew2],
                                     (const float*)d_in[I_eb2], hbuf);
  k_compact<<<(NE + 255) / 256, 256, 0, stream>>>(EI, NT, counter, src_act,
                                                  dst_act, eid_act);
  for (int l = 0; l < 3; ++l) {
    k_att<<<640, 256, 0, stream>>>(hbuf, EA, src_act, dst_act, eid_act, counter,
                                   aw1t[l], (const float*)d_in[I_ab1[l]],
                                   (const float*)d_in[I_aw2[l]],
                                   (const float*)d_in[I_ab2[l]], a_ws, amax);
    k_msg<<<640, 256, 0, stream>>>(hbuf, EA, src_act, dst_act, eid_act, counter,
                                   mw1t[l], (const float*)d_in[I_mb1[l]],
                                   mw2t[l], (const float*)d_in[I_mb2[l]],
                                   a_ws, amax, denom, aggr);
    k_node<<<256, 256, 0, stream>>>(hbuf, aggr, denom, amax,
                                    (const float*)d_in[I_lg[l]],
                                    (const float*)d_in[I_lb[l]]);
  }
  k_head<<<128, 256, 0, stream>>>(hbuf, HY, (const float*)d_in[I_hw1],
                                  (const float*)d_in[I_hb1],
                                  (const float*)d_in[I_hw2],
                                  (const float*)d_in[I_hb2], (float*)d_out);
}
